// MoEBlock_10883447128124
// MI455X (gfx1250) — matrix operand order repaired and checked by reading, not compiled, not run on hardware
//
#include <hip/hip_runtime.h>

// ---------------- problem constants (match reference) ----------------
#define D      1024
#define E_EXP  8
#define K_TOP  2
#define R_LORA 16
#define N_TOK  8192          // B*S = 4*2048
#define H_DIM  4096          // 4*D
#define SCALING 2.0f         // ALPHA / R = 32/16
#define NK_TOT (N_TOK * K_TOP)

// ---------------- WMMA types ----------------
typedef __attribute__((ext_vector_type(16))) __bf16 v16bf;
typedef __attribute__((ext_vector_type(2)))  __bf16 v2bf;
typedef __attribute__((ext_vector_type(8)))  float  v8f;
struct BFrag { unsigned short u[16]; };   // 32 bytes == v16bf

// native bf16 converts
__device__ __forceinline__ unsigned short f2bf(float f) {
    __bf16 b = (__bf16)f;
    return __builtin_bit_cast(unsigned short, b);
}
// packs to a single v_cvt_pk_bf16_f32
__device__ __forceinline__ unsigned int f2bf2(float lo, float hi) {
    v2bf v; v.x = (__bf16)lo; v.y = (__bf16)hi;
    return __builtin_bit_cast(unsigned int, v);
}

// ---- CDNA5 async global->LDS copy (ASYNCcnt-tracked), via inline asm ----
// VDST = per-lane LDS byte address, VADDR = per-lane 64-bit global address.
__device__ __forceinline__ void async_ld_b128(unsigned lds_addr, const void* gaddr) {
    asm volatile("global_load_async_to_lds_b128 %0, %1, off"
                 :: "v"(lds_addr), "v"(gaddr) : "memory");
}
__device__ __forceinline__ void wait_async0() {
    asm volatile("s_wait_asynccnt 0" ::: "memory");
}
// low 32 bits of a flat shared pointer == LDS byte offset (LDS aperture)
__device__ __forceinline__ unsigned lds_off32(const void* p) {
    return (unsigned)(size_t)p;
}

// =====================================================================
// 0) zero output (it is atomically accumulated) + routing counters
// =====================================================================
__global__ void k_zero(float* __restrict__ out, int n, int* __restrict__ counts) {
    int i = blockIdx.x * blockDim.x + threadIdx.x;
    if (i < n) out[i] = 0.0f;
    if (i < E_EXP) counts[i] = 0;
}

// =====================================================================
// 1) effective gate weight: w_eff[e,d] = base[e,d] + SCALING*(A@B)[d,e]
// =====================================================================
__global__ void k_weff(const float* __restrict__ base,
                       const float* __restrict__ lA,   // [D,R]
                       const float* __restrict__ lB,   // [R,E]
                       float* __restrict__ weff) {     // [E,D]
    int i = blockIdx.x * blockDim.x + threadIdx.x;
    if (i >= E_EXP * D) return;
    int e = i / D, d = i % D;
    float acc = 0.0f;
#pragma unroll
    for (int r = 0; r < R_LORA; ++r) acc += lA[d * R_LORA + r] * lB[r * E_EXP + e];
    weff[i] = base[i] + SCALING * acc;
}

// =====================================================================
// 2) task-emb add + gating + top-2 softmax + per-expert compaction
//    one wave32 per token
// =====================================================================
__global__ void k_gate(const float* __restrict__ x,
                       const int*   __restrict__ task,
                       const float* __restrict__ temb,
                       const float* __restrict__ weff,
                       unsigned short* __restrict__ xbf,   // [N,D] bf16 x_flat
                       int*   __restrict__ counts,
                       int*   __restrict__ rowTok,         // [E,N]
                       float* __restrict__ rowW) {         // [E,N]
    const int wave = threadIdx.x >> 5;
    const int lane = threadIdx.x & 31;
    const int n = blockIdx.x * 8 + wave;

    const float* xr = x + (size_t)n * D;
    const float* te = temb + (size_t)task[n] * D;

    float logit[E_EXP];
#pragma unroll
    for (int e = 0; e < E_EXP; ++e) logit[e] = 0.0f;

    for (int c = 0; c < D / 32; ++c) {
        int d = c * 32 + lane;
        float v = xr[d] + te[d];
        xbf[(size_t)n * D + d] = f2bf(v);
#pragma unroll
        for (int e = 0; e < E_EXP; ++e) logit[e] += v * weff[e * D + d];
    }
#pragma unroll
    for (int e = 0; e < E_EXP; ++e) {
        float v = logit[e];
        for (int off = 16; off > 0; off >>= 1) v += __shfl_xor(v, off, 32);
        logit[e] = v;
    }
    if (lane == 0) {
        int i0 = 0; float v0 = logit[0];
#pragma unroll
        for (int e = 1; e < E_EXP; ++e) if (logit[e] > v0) { v0 = logit[e]; i0 = e; }
        int i1 = (i0 == 0) ? 1 : 0; float v1 = logit[i1];
#pragma unroll
        for (int e = 0; e < E_EXP; ++e)
            if (e != i0 && logit[e] > v1) { v1 = logit[e]; i1 = e; }
        float w0 = 1.0f / (1.0f + __expf(v1 - v0));   // softmax over {v0,v1}
        float w1 = 1.0f - w0;
        int s0 = atomicAdd(&counts[i0], 1);
        rowTok[i0 * N_TOK + s0] = n;  rowW[i0 * N_TOK + s0] = w0;
        int s1 = atomicAdd(&counts[i1], 1);
        rowTok[i1 * N_TOK + s1] = n;  rowW[i1 * N_TOK + s1] = w1;
    }
}

// =====================================================================
// 3) exclusive prefix of counts
// =====================================================================
__global__ void k_offs(const int* __restrict__ counts, int* __restrict__ offs) {
    if (threadIdx.x == 0) {
        int a = 0;
        for (int e = 0; e < E_EXP; ++e) { offs[e] = a; a += counts[e]; }
        offs[E_EXP] = a;
    }
}

// =====================================================================
// WMMA fragment layouts (wave32, per CDNA5 ISA):
//  A 16x32 bf16: lane<16 row M=lane holds K 0..7 (v0..3) & 16..23 (v4..7);
//                lane>=16 same M, K shifted +8.
//  B 32x16 bf16: lane%16 = col N, element t -> K = t + 8*(t>=8) + 8*(lane>=16) (as A).
//  C 16x16 f32 : element r -> M = r + 8*(lane>=16), N = lane%16.
// =====================================================================
#define ASTR 40   // LDS row stride in bf16: 80 B = 5*16 B -> rows 16B-aligned

// =====================================================================
// 4) GEMM1: h[slot, :] = gather(x_bf)[rows] @ w1[e]^T + b1[e]   (bf16 out)
//    block tile 128(M) x 128(H-cols), K-step 32, LDS ping-pong.
//    A tile: async global->LDS (no VGPR staging).  B tile: reg-staged
//    fp32->bf16 convert, issued one iteration ahead.
// =====================================================================
__global__ void __launch_bounds__(256, 1)
k_gemm1(const unsigned short* __restrict__ xbf,
        const int*   __restrict__ rowTok,
        const int*   __restrict__ counts,
        const int*   __restrict__ offs,
        const float* __restrict__ w1,     // [E,H,D]
        const float* __restrict__ b1,     // [E,H]
        unsigned short* __restrict__ hbuf) { // [N*K, H] bf16
    const int e   = blockIdx.z;
    const int cnt = counts[e];
    const int m0  = blockIdx.x * 128;
    if (m0 >= cnt) return;                 // uniform over the block
    const int n0    = blockIdx.y * 128;
    const int offsE = offs[e];

    __shared__ unsigned short As[2][128 * ASTR];
    __shared__ unsigned short Bs[2][128 * ASTR];

    const int w    = threadIdx.x >> 5;
    const int lane = threadIdx.x & 31;
    const int lm   = lane & 15;
    const int hi   = lane >> 4;            // 0 or 1
    const int mbase = (w >> 2) * 64;       // 2 wave-rows
    const int nbase = (w & 3) * 32;        // 4 wave-cols

    // ---- staging slots ----
    const float* bB = w1 + (size_t)e * H_DIM * D + (size_t)n0 * D;  // uniform
    int aG[2], aL[2];
#pragma unroll
    for (int p = 0; p < 2; ++p) {
        int slot = threadIdx.x + p * 256;          // 512 slots of 8 bf16
        int row = slot >> 2, col = (slot & 3) * 8;
        int tok = (m0 + row < cnt) ? rowTok[e * N_TOK + m0 + row] : 0;
        aG[p] = tok * D + col;                     // < 2^31
        aL[p] = row * ASTR + col;                  // 16B-aligned (col%8==0)
    }
    int bG[4], bL[4];
#pragma unroll
    for (int p = 0; p < 4; ++p) {
        int slot = threadIdx.x + p * 256;          // 1024 slots of 4 f32
        int row = slot >> 3, col = (slot & 7) * 4;
        bG[p] = row * D + col;
        bL[p] = row * ASTR + col;                  // 8B-aligned (col%4==0)
    }

    // A: async DMA, no registers held
    auto asyncA = [&](int buf, int k0) {
#pragma unroll
        for (int p = 0; p < 2; ++p)
            async_ld_b128(lds_off32(&As[buf][aL[p]]), xbf + aG[p] + k0);
    };
    // B: register-staged with convert
    float4 rb[4];
    auto loadB = [&](int k0) {
#pragma unroll
        for (int p = 0; p < 4; ++p) rb[p] = *(const float4*)(bB + bG[p] + k0);
    };
    auto storeB = [&](int buf) {
#pragma unroll
        for (int p = 0; p < 4; ++p) {
            uint2 pk;
            pk.x = f2bf2(rb[p].x, rb[p].y);
            pk.y = f2bf2(rb[p].z, rb[p].w);
            *(uint2*)&Bs[buf][bL[p]] = pk;                          // b64 store
        }
    };

    v8f acc[4][2];
#pragma unroll
    for (int mi = 0; mi < 4; ++mi)
#pragma unroll
        for (int ni = 0; ni < 2; ++ni)
            acc[mi][ni] = (v8f){0.f,0.f,0.f,0.f,0.f,0.f,0.f,0.f};

    const int NK = D / 32;
    asyncA(0, 0);
    loadB(0);
    storeB(0);
    loadB(32);

#pragma unroll 2
    for (int k = 0; k < NK; ++k) {
        wait_async0();                             // my async tile-k LDS writes done
        __syncthreads();                           // everyone's writes visible
        if (k + 1 < NK) {
            asyncA((k + 1) & 1, (k + 1) * 32);     // DMA next A tile, overlaps WMMA
            storeB((k + 1) & 1);                   // waits B loads from last iter
            if (k + 2 < NK) loadB((k + 2) * 32);   // issue ahead
        }
        const int buf = k & 1;

        v16bf afr[4], bfr[2];
#pragma unroll
        for (int mi = 0; mi < 4; ++mi) {
            BFrag f;
            int row = mbase + mi * 16 + lm;
#pragma unroll
            for (int t = 0; t < 16; ++t) {
                int vg = t >> 1;
                int kb = (vg < 4) ? (vg * 2) : (16 + (vg - 4) * 2);
                f.u[t] = As[buf][row * ASTR + kb + (t & 1) + hi * 8];
            }
            afr[mi] = __builtin_bit_cast(v16bf, f);
        }
#pragma unroll
        for (int ni = 0; ni < 2; ++ni) {
            BFrag f;
            int col = nbase + ni * 16 + lm;
#pragma unroll
            for (int t = 0; t < 16; ++t) f.u[t] = Bs[buf][col * ASTR + t + (t >> 3) * 8 + hi * 8];
            bfr[ni] = __builtin_bit_cast(v16bf, f);
        }
#pragma unroll
        for (int mi = 0; mi < 4; ++mi)
#pragma unroll
            for (int ni = 0; ni < 2; ++ni)
                acc[mi][ni] = __builtin_amdgcn_wmma_f32_16x16x32_bf16(
                    false, afr[mi], false, bfr[ni], (short)0, acc[mi][ni], false, false);
    }

    // ---- epilogue: +b1, bf16 store into compacted h buffer ----
#pragma unroll
    for (int mi = 0; mi < 4; ++mi)
#pragma unroll
        for (int r = 0; r < 8; ++r) {
            int m = mbase + mi * 16 + r + hi * 8;
            if (m0 + m < cnt) {
#pragma unroll
                for (int ni = 0; ni < 2; ++ni) {
                    int gc = n0 + nbase + ni * 16 + lm;
                    float val = acc[mi][ni][r] + b1[(size_t)e * H_DIM + gc];
                    hbuf[(size_t)(offsE + m0 + m) * H_DIM + gc] = f2bf(val);
                }
            }
        }
}

// =====================================================================
// 5) GEMM2: g = h@wg^T+bg ; v = h@wv^T+bv ; out[token] += cw*silu(g)*v
//    block tile 128(M) x 64(D-cols), K-step 32 over H, shared A tile,
//    A async global->LDS, B reg-staged, same ping-pong.
// =====================================================================
__global__ void __launch_bounds__(256, 1)
k_gemm2(const unsigned short* __restrict__ hbuf,   // [N*K, H]
        const int*   __restrict__ rowTok,
        const float* __restrict__ rowW,
        const int*   __restrict__ counts,
        const int*   __restrict__ offs,
        const float* __restrict__ wg,     // [E,D,H]
        const float* __restrict__ bg,     // [E,D]
        const float* __restrict__ wv,     // [E,D,H]
        const float* __restrict__ bv,     // [E,D]
        float* __restrict__ out) {        // [N,D]
    const int e   = blockIdx.z;
    const int cnt = counts[e];
    const int m0  = blockIdx.x * 128;
    if (m0 >= cnt) return;
    const int d0    = blockIdx.y * 64;
    const int offsE = offs[e];

    __shared__ unsigned short As[2][128 * ASTR];
    __shared__ unsigned short Bgs[2][64 * ASTR];
    __shared__ unsigned short Bvs[2][64 * ASTR];

    const int w    = threadIdx.x >> 5;
    const int lane = threadIdx.x & 31;
    const int lm   = lane & 15;
    const int hi   = lane >> 4;
    const int mbase = (w >> 1) * 32;      // 4 wave-rows
    const int nbase = (w & 1) * 32;       // 2 wave-cols

    const float* gB = wg + (size_t)e * D * H_DIM + (size_t)d0 * H_DIM;  // uniform
    const float* vB = wv + (size_t)e * D * H_DIM + (size_t)d0 * H_DIM;  // uniform
    int aG[2], aL[2];
#pragma unroll
    for (int p = 0; p < 2; ++p) {
        int slot = threadIdx.x + p * 256;
        int row = slot >> 2, col = (slot & 3) * 8;
        int hrow = offsE + m0 + row;
        if (hrow > NK_TOT - 1) hrow = NK_TOT - 1;
        aG[p] = hrow * H_DIM + col;                // < 2^31
        aL[p] = row * ASTR + col;
    }
    int bG[2], bL[2];
#pragma unroll
    for (int p = 0; p < 2; ++p) {
        int slot = threadIdx.x + p * 256;          // 512 slots of 4 f32
        int row = slot >> 3, col = (slot & 7) * 4;
        bG[p] = row * H_DIM + col;
        bL[p] = row * ASTR + col;
    }

    auto asyncA = [&](int buf, int k0) {
#pragma unroll
        for (int p = 0; p < 2; ++p)
            async_ld_b128(lds_off32(&As[buf][aL[p]]), hbuf + aG[p] + k0);
    };
    float4 rg[2], rv[2];
    auto loadB = [&](int k0) {
#pragma unroll
        for (int p = 0; p < 2; ++p) {
            rg[p] = *(const float4*)(gB + bG[p] + k0);
            rv[p] = *(const float4*)(vB + bG[p] + k0);
        }
    };
    auto storeB = [&](int buf) {
#pragma unroll
        for (int p = 0; p < 2; ++p) {
            uint2 pg, pv;
            pg.x = f2bf2(rg[p].x, rg[p].y);
            pg.y = f2bf2(rg[p].z, rg[p].w);
            *(uint2*)&Bgs[buf][bL[p]] = pg;
            pv.x = f2bf2(rv[p].x, rv[p].y);
            pv.y = f2bf2(rv[p].z, rv[p].w);
            *(uint2*)&Bvs[buf][bL[p]] = pv;
        }
    };

    v8f accg[2][2], accv[2][2];
#pragma unroll
    for (int mi = 0; mi < 2; ++mi)
#pragma unroll
        for (int ni = 0; ni < 2; ++ni) {
            accg[mi][ni] = (v8f){0.f,0.f,0.f,0.f,0.f,0.f,0.f,0.f};
            accv[mi][ni] = (v8f){0.f,0.f,0.f,0.f,0.f,0.f,0.f,0.f};
        }

    const int NK = H_DIM / 32;
    asyncA(0, 0);
    loadB(0);
    storeB(0);
    loadB(32);

#pragma unroll 2
    for (int k = 0; k < NK; ++k) {
        wait_async0();
        __syncthreads();
        if (k + 1 < NK) {
            asyncA((k + 1) & 1, (k + 1) * 32);
            storeB((k + 1) & 1);
            if (k + 2 < NK) loadB((k + 2) * 32);
        }
        const int buf = k & 1;

        v16bf afr[2], bgf[2], bvf[2];
#pragma unroll
        for (int mi = 0; mi < 2; ++mi) {
            BFrag f;
            int row = mbase + mi * 16 + lm;
#pragma unroll
            for (int t = 0; t < 16; ++t) {
                int vg = t >> 1;
                int kb = (vg < 4) ? (vg * 2) : (16 + (vg - 4) * 2);
                f.u[t] = As[buf][row * ASTR + kb + (t & 1) + hi * 8];
            }
            afr[mi] = __builtin_bit_cast(v16bf, f);
        }
#pragma unroll
        for (int ni = 0; ni < 2; ++ni) {
            BFrag fg, fv;
            int col = nbase + ni * 16 + lm;
#pragma unroll
            for (int t = 0; t < 16; ++t) {
                fg.u[t] = Bgs[buf][col * ASTR + t + (t >> 3) * 8 + hi * 8];
                fv.u[t] = Bvs[buf][col * ASTR + t + (t >> 3) * 8 + hi * 8];
            }
            bgf[ni] = __builtin_bit_cast(v16bf, fg);
            bvf[ni] = __builtin_bit_cast(v16bf, fv);
        }
#pragma unroll
        for (int mi = 0; mi < 2; ++mi)
#pragma unroll
            for (int ni = 0; ni < 2; ++ni) {
                accg[mi][ni] = __builtin_amdgcn_wmma_f32_16x16x32_bf16(
                    false, afr[mi], false, bgf[ni], (short)0, accg[mi][ni], false, false);
                accv[mi][ni] = __builtin_amdgcn_wmma_f32_16x16x32_bf16(
                    false, afr[mi], false, bvf[ni], (short)0, accv[mi][ni], false, false);
            }
    }

    // ---- epilogue: SiLU(g+bg)*(v+bv)*cw, scatter-add into out ----
#pragma unroll
    for (int mi = 0; mi < 2; ++mi)
#pragma unroll
        for (int r = 0; r < 8; ++r) {
            int m = mbase + mi * 16 + r + hi * 8;
            if (m0 + m < cnt) {
                int   tok = rowTok[e * N_TOK + m0 + m];
                float cw  = rowW[e * N_TOK + m0 + m];
#pragma unroll
                for (int ni = 0; ni < 2; ++ni) {
                    int gc = d0 + nbase + ni * 16 + lm;
                    float g  = accg[mi][ni][r] + bg[(size_t)e * D + gc];
                    float vv = accv[mi][ni][r] + bv[(size_t)e * D + gc];
                    float sig = 1.0f / (1.0f + __expf(-g));
                    atomicAdd(&out[(size_t)tok * D + gc], g * sig * vv * cw);
                }
            }
        }
}

// =====================================================================
// launcher
// =====================================================================
extern "C" void kernel_launch(void* const* d_in, const int* in_sizes, int n_in,
                              void* d_out, int out_size, void* d_ws, size_t ws_size,
                              hipStream_t stream) {
    const float* x    = (const float*)d_in[0];
    const int*   task = (const int*)  d_in[1];
    const float* temb = (const float*)d_in[2];
    const float* bgw  = (const float*)d_in[3];
    const float* lA   = (const float*)d_in[4];
    const float* lB   = (const float*)d_in[5];
    const float* w1   = (const float*)d_in[6];
    const float* b1   = (const float*)d_in[7];
    const float* wg   = (const float*)d_in[8];
    const float* bgb  = (const float*)d_in[9];
    const float* wv   = (const float*)d_in[10];
    const float* bvb  = (const float*)d_in[11];
    float* out = (float*)d_out;

    char* p = (char*)d_ws;
    auto alloc = [&](size_t bytes) -> char* {
        char* r = p; p += (bytes + 255) & ~(size_t)255; return r;
    };
    float*          weff   = (float*)alloc((size_t)E_EXP * D * 4);
    int*            counts = (int*)  alloc(E_EXP * 4);
    int*            offs   = (int*)  alloc((E_EXP + 1) * 4);
    int*            rowTok = (int*)  alloc((size_t)E_EXP * N_TOK * 4);
    float*          rowW   = (float*)alloc((size_t)E_EXP * N_TOK * 4);
    unsigned short* xbf    = (unsigned short*)alloc((size_t)N_TOK * D * 2);
    unsigned short* hbuf   = (unsigned short*)alloc((size_t)NK_TOT * H_DIM * 2);

    k_zero<<<(N_TOK * D + 255) / 256, 256, 0, stream>>>(out, N_TOK * D, counts);
    k_weff<<<(E_EXP * D + 255) / 256, 256, 0, stream>>>(bgw, lA, lB, weff);
    k_gate<<<N_TOK / 8, 256, 0, stream>>>(x, task, temb, weff, xbf, counts, rowTok, rowW);
    k_offs<<<1, 32, 0, stream>>>(counts, offs);

    dim3 g1(N_TOK / 128, H_DIM / 128, E_EXP);
    k_gemm1<<<g1, 256, 0, stream>>>(xbf, rowTok, counts, offs, w1, b1, hbuf);

    dim3 g2(N_TOK / 128, D / 64, E_EXP);
    k_gemm2<<<g2, 256, 0, stream>>>(hbuf, rowTok, rowW, counts, offs, wg, bgb, wv, bvb, out);
}
